// KoopmanMlpExtractor_4458176053291
// MI455X (gfx1250) — matrix operand order repaired and checked by reading, not compiled, not run on hardware
//
#include <hip/hip_runtime.h>
#include <hip/hip_bf16.h>

typedef _Float16 h16;
typedef __attribute__((ext_vector_type(16))) _Float16 v16h;
typedef __attribute__((ext_vector_type(8)))  _Float16 v8h;
typedef __attribute__((ext_vector_type(4)))  _Float16 v4h;
typedef __attribute__((ext_vector_type(8)))  float    v8f;
typedef __attribute__((ext_vector_type(4)))  float    v4f;
typedef unsigned int u32x4 __attribute__((ext_vector_type(4)));
typedef int          i32x8 __attribute__((ext_vector_type(8)));
typedef int          i32x4 __attribute__((ext_vector_type(4)));

#define D_IN   512
#define D_OUT  256
#define HID    64
#define BATCH  65536
#define MTILE  128
#define XS     520   // padded f16 stride for x tile (rows shift 4 LDS banks)
#define WS     520   // padded f16 stride for weight panel (matches TDM pad)
#define NPANEL 25    // 1 chooser + 16 critic + 8 actor weight panels

union V16U { v16h v; v8h h[2]; };

__device__ __forceinline__ v8f vzero() {
  v8f z;
#pragma unroll
  for (int i = 0; i < 8; ++i) z[i] = 0.0f;
  return z;
}

// A-operand (16x32 f16): lanes 0-15 row=lane, K{0..7,16..23}; lanes 16-31 K{8..15,24..31}.
__device__ __forceinline__ v16h ld_afrag(const h16* base, int lane) {
  const h16* p = base + (lane & 15) * XS + ((lane & 16) ? 8 : 0);
  V16U u;
  u.h[0] = *(const v8h*)p;
  u.h[1] = *(const v8h*)(p + 16);
  return u.v;
}

// B-operand (32x16 f16): lane holds col=lane&15; lanes 0-15 K{0..7,16..23}; lanes 16-31 K{8..15,24..31}.
__device__ __forceinline__ v16h ld_bfrag(const h16* base, int lane) {
  const h16* p = base + (lane & 15) * WS + ((lane & 16) ? 8 : 0);
  V16U u;
  u.h[0] = *(const v8h*)p;
  u.h[1] = *(const v8h*)(p + 16);
  return u.v;
}

#define WMMA(a, b, c) __builtin_amdgcn_wmma_f32_16x16x32_f16( \
    false, (a), false, (b), (short)0, (c), false, false)

// ---- TDM: DMA one 64x512 f16 weight panel global->LDS with 16B/row padding ----
// D# per CDNA5 ISA:
//   group0: count=1, lds_addr, global_addr[56:0], type=2
//   group1: data_size=2B, pad_enable, pad_interval=256 DWORDs, pad_amount=4 DWORDs,
//           tensor_dim0=512, tensor_dim1=64, tile_dim0=512, tile_dim1=64,
//           tensor_dim0_stride=512  (effective LDS row stride = 1040 B = WS halves)
__device__ __forceinline__ void tdm_load_panel(const h16* gsrc, unsigned lds_byte_off) {
  unsigned long long ga = (unsigned long long)(uintptr_t)gsrc;
  u32x4 g0;
  g0[0] = 1u;                                            // count=1 (valid user desc)
  g0[1] = lds_byte_off;                                  // lds_addr
  g0[2] = (unsigned)(ga & 0xFFFFFFFFu);                  // global_addr[31:0]
  g0[3] = (unsigned)((ga >> 32) & 0x1FFFFFFu) | (2u << 30); // global_addr[56:32] | type=2

  i32x8 g1;
  g1[0] = (int)((1u << 16)      // data_size = 2 bytes
              | (1u << 20)      // pad_enable
              | (7u << 22)      // pad_interval: 256 DWORDs (one 1024B row)
              | (3u << 25));    // pad_amount:   4 DWORDs (16B)
  g1[1] = (int)(512u << 16);                      // tensor_dim0[15:0] in [31:16]
  g1[2] = (int)((512u >> 16) | (64u << 16));      // tensor_dim0[31:16] | tensor_dim1[15:0]
  g1[3] = (int)((64u >> 16) | (512u << 16));      // tensor_dim1[31:16] | tile_dim0=512
  g1[4] = (int)64u;                               // tile_dim1=64, tile_dim2=0
  g1[5] = (int)512;                               // tensor_dim0_stride[31:0]
  g1[6] = 0;                                      // stride hi | tensor_dim1_stride lo
  g1[7] = 0;

  i32x4 z4 = {0, 0, 0, 0};
#if defined(__clang_major__) && (__clang_major__ >= 23)
  i32x8 z8 = {0, 0, 0, 0, 0, 0, 0, 0};
  __builtin_amdgcn_tensor_load_to_lds(g0, g1, z4, z4, z8, 0);
#else
  __builtin_amdgcn_tensor_load_to_lds(g0, g1, z4, z4, 0);
#endif
}

// ---------------- pre-pass: fp32 -> f16 weight conversion ----------------
__global__ void cvt_f16_kernel(const float* __restrict__ s, h16* __restrict__ d, int n) {
  int i = blockIdx.x * 256 + threadIdx.x;
  if (i < n) d[i] = (h16)s[i];
}

// At[j][k] = (f16) A[k][j]  (512x512), tiled transpose through LDS
__global__ void transpose_cvt512_kernel(const float* __restrict__ A, h16* __restrict__ At) {
  __shared__ float t[32][33];
  int x0 = blockIdx.x * 32, y0 = blockIdx.y * 32;
  int lx = threadIdx.x, ly = threadIdx.y; // (32, 8)
#pragma unroll
  for (int i = 0; i < 32; i += 8)
    t[ly + i][lx] = A[(size_t)(y0 + ly + i) * 512 + x0 + lx];
  __syncthreads();
#pragma unroll
  for (int i = 0; i < 32; i += 8)
    At[(size_t)(x0 + ly + i) * 512 + y0 + lx] = (h16)t[lx][ly + i];
}

// ---------------- fused main kernel ----------------
__global__ __launch_bounds__(256) void fused_kernel(
    const float* __restrict__ x,
    const h16* __restrict__ W1h, const h16* __restrict__ W2h,
    const h16* __restrict__ Wc1h,
    const h16* __restrict__ A1t, const h16* __restrict__ A2t,
    const float* __restrict__ b1, const float* __restrict__ c1,
    const float* __restrict__ b2, const float* __restrict__ c2,
    const float* __restrict__ bc1, const float* __restrict__ Wc2,
    const float* __restrict__ bc2,
    float* __restrict__ actor, float* __restrict__ critic)
{
  __shared__ alignas(16) h16 xs[MTILE * XS];      // 133,120 B : x tile f16
  __shared__ alignas(16) h16 wpb[2][64 * WS];     // 133,120 B : double-buffered weight panel
  __shared__ float Hs[MTILE * 68];                //  34,816 B : chooser hidden
  __shared__ float sig[MTILE * 2];
  __shared__ float qz[2][MTILE];
  __shared__ float xbp[2][MTILE][2];

  const int tid  = threadIdx.x;
  const int lane = tid & 31;
  const int wid  = tid >> 5;
  const int R0   = wid * 16;                 // wave's 16-row M tile
  const int col0 = lane & 15;
  const int rowh = (lane & 16) ? 8 : 0;      // D-tile row half per ISA layout
  const long long row0 = (long long)blockIdx.x * MTILE;

  // panel schedule: s=0 -> Wc1; s=1..16 -> A1t/A2t panels; s=17..24 -> W1/W2 interleaved
  auto panel_ptr = [&](int s) -> const h16* {
    if (s == 0) return Wc1h;
    if (s <= 16) {
      int i = s - 1;
      return (i < 8) ? (A1t + (size_t)i * 64 * D_IN) : (A2t + (size_t)(i - 8) * 64 * D_IN);
    }
    int i = s - 17;
    return (((i & 1) ? W2h : W1h) + (size_t)(i >> 1) * 64 * D_IN);
  };
  unsigned wpoff[2];
  wpoff[0] = (unsigned)(uintptr_t)&wpb[0][0];
  wpoff[1] = (unsigned)(uintptr_t)&wpb[1][0];

  // prologue: wave 0 primes the 2-deep TDM pipeline (overlaps x-tile load below)
  if (wid == 0) {
    tdm_load_panel(panel_ptr(0), wpoff[0]);
    tdm_load_panel(panel_ptr(1), wpoff[1]);
  }

  // ---- phase 1: load x tile, convert fp32 -> f16 into LDS ----
  const float* xg = x + row0 * D_IN;
  for (int t = tid; t < MTILE * (D_IN / 4); t += 256) {
    int r  = t >> 7;              // D_IN/4 = 128 float4 chunks per row
    int kc = (t & 127) << 2;
    v4f v = *(const v4f*)(xg + r * D_IN + kc);
    v4h h; h.x = (h16)v.x; h.y = (h16)v.y; h.z = (h16)v.z; h.w = (h16)v.w;
    *(v4h*)(xs + r * XS + kc) = h;
  }
  __syncthreads();

  // ---- phase 2: chooser  H = relu(x @ Wc1^T + bc1)   (consume panel s=0) ----
  if (wid == 0) __builtin_amdgcn_s_wait_tensorcnt(1);   // panel 0 done (1 stays in flight)
  __syncthreads();
  {
    const h16* wp = wpb[0];
    v8f acc[4];
#pragma unroll
    for (int i = 0; i < 4; ++i) acc[i] = vzero();
    for (int k = 0; k < D_IN; k += 32) {
      v16h a = ld_afrag(xs + R0 * XS + k, lane);
#pragma unroll
      for (int nt = 0; nt < 4; ++nt) {
        v16h b = ld_bfrag(wp + nt * 16 * WS + k, lane);
        acc[nt] = WMMA(a, b, acc[nt]);
      }
    }
#pragma unroll
    for (int nt = 0; nt < 4; ++nt) {
      float bias = bc1[nt * 16 + col0];
#pragma unroll
      for (int r = 0; r < 8; ++r) {
        float hv = acc[nt][r] + bias;
        Hs[(R0 + rowh + r) * 68 + nt * 16 + col0] = hv > 0.0f ? hv : 0.0f;
      }
    }
  }
  __syncthreads();                                      // all waves done with wpb[0]
  if (wid == 0) tdm_load_panel(panel_ptr(2), wpoff[0]); // refill buffer 0

  // ---- phase 3: logits + sigmoid, and x.b partials (no panel use; overlaps TDM) ----
  {
    int row = tid >> 1, hh = tid & 1;
    float lg = bc2[hh];
    const float* w2r = Wc2 + hh * HID;
    for (int k = 0; k < HID; ++k) lg += Hs[row * 68 + k] * w2r[k];
    sig[row * 2 + hh] = 1.0f / (1.0f + __expf(-lg));

    float s1 = 0.0f, s2 = 0.0f;
    const h16* xr = xs + row * XS + hh * 256;
    const float* bb1 = b1 + hh * 256;
    const float* bb2 = b2 + hh * 256;
    for (int k = 0; k < 256; ++k) {
      float xv = (float)xr[k];
      s1 += xv * bb1[k];
      s2 += xv * bb2[k];
    }
    xbp[0][row][hh] = s1;
    xbp[1][row][hh] = s2;
  }
  __syncthreads();

  float s0r[8], s1r[8];
#pragma unroll
  for (int r = 0; r < 8; ++r) {
    int row = R0 + rowh + r;
    s0r[r] = sig[row * 2];
    s1r[r] = sig[row * 2 + 1];
  }

  // ---- phase 4: critic  q_m = sum_j (x @ A_m)[i,j] * x[i,j]  (panels s=1..16) ----
  float qacc[2][8];
#pragma unroll
  for (int m = 0; m < 2; ++m)
#pragma unroll
    for (int r = 0; r < 8; ++r) qacc[m][r] = 0.0f;

  for (int m = 0; m < 2; ++m) {
    for (int jp = 0; jp < 8; ++jp) {
      int s = 1 + m * 8 + jp;
      if (wid == 0) __builtin_amdgcn_s_wait_tensorcnt(1);
      __syncthreads();
      const h16* wp = wpb[s & 1];
      v8f acc[4];
#pragma unroll
      for (int i = 0; i < 4; ++i) acc[i] = vzero();
      for (int k = 0; k < D_IN; k += 32) {
        v16h a = ld_afrag(xs + R0 * XS + k, lane);
#pragma unroll
        for (int nt = 0; nt < 4; ++nt) {
          v16h b = ld_bfrag(wp + nt * 16 * WS + k, lane);
          acc[nt] = WMMA(a, b, acc[nt]);
        }
      }
#pragma unroll
      for (int nt = 0; nt < 4; ++nt) {
#pragma unroll
        for (int r = 0; r < 8; ++r) {
          float xv = (float)xs[(R0 + rowh + r) * XS + jp * 64 + nt * 16 + col0];
          qacc[m][r] += acc[nt][r] * xv;
        }
      }
      __syncthreads();                          // all waves done with this buffer
      if (wid == 0) tdm_load_panel(panel_ptr(s + 2), wpoff[(s + 2) & 1]); // s+2 <= 18 here
    }
  }
  // reduce across the 16 lanes holding the same rows
#pragma unroll
  for (int r = 0; r < 8; ++r) {
    float v0 = qacc[0][r], v1 = qacc[1][r];
    for (int off = 8; off >= 1; off >>= 1) {
      v0 += __shfl_xor(v0, off, 16);
      v1 += __shfl_xor(v1, off, 16);
    }
    if (col0 == 0) {
      qz[0][R0 + rowh + r] = v0;
      qz[1][R0 + rowh + r] = v1;
    }
  }
  __syncthreads();
  if (tid < MTILE) {
    int row = tid;
    float q1 = qz[0][row] + xbp[0][row][0] + xbp[0][row][1] + c1[0];
    float q2 = qz[1][row] + xbp[1][row][0] + xbp[1][row][1] + c2[0];
    critic[row0 + row] = sig[row * 2] * q1 + sig[row * 2 + 1] * q2;
  }

  // ---- phase 5: actor (panels s=17..24: W1p0,W2p0,W1p1,W2p1,...) ----
  v8f acc1[4], acc2[4];
  for (int s = 17; s < NPANEL; ++s) {
    int i = s - 17, np = i >> 1;
    if (wid == 0) {
      if (s == NPANEL - 1) __builtin_amdgcn_s_wait_tensorcnt(0);  // nothing newer in flight
      else                 __builtin_amdgcn_s_wait_tensorcnt(1);
    }
    __syncthreads();
    const h16* wp = wpb[s & 1];
    if ((i & 1) == 0) {
#pragma unroll
      for (int t = 0; t < 4; ++t) acc1[t] = vzero();
      for (int k = 0; k < D_IN; k += 32) {
        v16h a = ld_afrag(xs + R0 * XS + k, lane);
#pragma unroll
        for (int nt = 0; nt < 4; ++nt) {
          v16h b = ld_bfrag(wp + nt * 16 * WS + k, lane);
          acc1[nt] = WMMA(a, b, acc1[nt]);
        }
      }
    } else {
#pragma unroll
      for (int t = 0; t < 4; ++t) acc2[t] = vzero();
      for (int k = 0; k < D_IN; k += 32) {
        v16h a = ld_afrag(xs + R0 * XS + k, lane);
#pragma unroll
        for (int nt = 0; nt < 4; ++nt) {
          v16h b = ld_bfrag(wp + nt * 16 * WS + k, lane);
          acc2[nt] = WMMA(a, b, acc2[nt]);
        }
      }
#pragma unroll
      for (int nt = 0; nt < 4; ++nt) {
#pragma unroll
        for (int r = 0; r < 8; ++r) {
          int row = R0 + rowh + r;
          actor[(row0 + row) * D_OUT + np * 64 + nt * 16 + col0] =
              s0r[r] * acc1[nt][r] + s1r[r] * acc2[nt][r];
        }
      }
    }
    __syncthreads();
    if (wid == 0 && (s + 2) < NPANEL) tdm_load_panel(panel_ptr(s + 2), wpoff[(s + 2) & 1]);
  }
}

extern "C" void kernel_launch(void* const* d_in, const int* in_sizes, int n_in,
                              void* d_out, int out_size, void* d_ws, size_t ws_size,
                              hipStream_t stream) {
  const float* x   = (const float*)d_in[0];
  const float* W1  = (const float*)d_in[1];
  const float* W2  = (const float*)d_in[2];
  const float* A1  = (const float*)d_in[3];
  const float* b1  = (const float*)d_in[4];
  const float* c1  = (const float*)d_in[5];
  const float* A2  = (const float*)d_in[6];
  const float* b2  = (const float*)d_in[7];
  const float* c2  = (const float*)d_in[8];
  const float* Wc1 = (const float*)d_in[9];
  const float* bc1 = (const float*)d_in[10];
  const float* Wc2 = (const float*)d_in[11];
  const float* bc2 = (const float*)d_in[12];

  float* actor  = (float*)d_out;
  float* critic = actor + (size_t)BATCH * D_OUT;

  // f16 weight scratch layout in d_ws (total 1,638,400 B)
  h16* W1h  = (h16*)d_ws;
  h16* W2h  = W1h  + (size_t)D_OUT * D_IN;
  h16* Wc1h = W2h  + (size_t)D_OUT * D_IN;
  h16* A1t  = Wc1h + (size_t)HID   * D_IN;
  h16* A2t  = A1t  + (size_t)D_IN  * D_IN;

  cvt_f16_kernel<<<(D_OUT * D_IN) / 256, 256, 0, stream>>>(W1, W1h, D_OUT * D_IN);
  cvt_f16_kernel<<<(D_OUT * D_IN) / 256, 256, 0, stream>>>(W2, W2h, D_OUT * D_IN);
  cvt_f16_kernel<<<(HID * D_IN) / 256, 256, 0, stream>>>(Wc1, Wc1h, HID * D_IN);
  dim3 tg(16, 16), tb(32, 8);
  transpose_cvt512_kernel<<<tg, tb, 0, stream>>>(A1, A1t);
  transpose_cvt512_kernel<<<tg, tb, 0, stream>>>(A2, A2t);

  fused_kernel<<<BATCH / MTILE, 256, 0, stream>>>(
      x, W1h, W2h, Wc1h, A1t, A2t, b1, c1, b2, c2, bc1, Wc2, bc2, actor, critic);
}
